// TripletLossModule_317827580073
// MI455X (gfx1250) — compile-verified, matrix operand order checked by reading, not run on hardware
//
#include <hip/hip_runtime.h>
#include <hip/hip_bf16.h>

// ---------------------------------------------------------------------------
// Batch-hard triplet loss for MI455X (gfx1250).
// Gram matrix via bf16x3 split-precision WMMA (v_wmma_f32_16x16x32_bf16).
// M=64 row blocking; A resident in LDS laid out [row][part][m][520] so ALL
// inner-loop DS accesses are immediate-offset off one per-lane base (<8.3KB
// span): no address VALU, no WMMA WAR-hazard NOPs. Row sq-norms/labels kept
// in LDS (not registers) to stay under 256 VGPRs. Term-major WMMA issue ->
// consecutive WMMAs hit independent accumulators.
// ---------------------------------------------------------------------------

typedef __attribute__((ext_vector_type(16))) __bf16         v16bf;
typedef __attribute__((ext_vector_type(8)))  __bf16         v8bf;
typedef __attribute__((ext_vector_type(8)))  float          v8f;
typedef __attribute__((ext_vector_type(4)))  int            i4;
typedef __attribute__((ext_vector_type(4)))  float          f4;
typedef __attribute__((ext_vector_type(2)))  float          f2;
typedef __attribute__((ext_vector_type(4)))  unsigned short us4;

#define BIGV    1.0e9f
#define MARGINV 1.0f
#define DCONST  512
#define MTILES  4
#define MROWS   (16 * MTILES)   // 64 rows per workgroup

// LDS A layout: [row(16)][part(2)][m(4)][KPAD] elements, padded row stride.
#define KPAD    520             // per-(row,part,m) K extent incl. pad
#define PSTRIDE (MTILES * KPAD) // part stride: 2080 elems (4160 B)
#define RSTRIDE (2 * PSTRIDE + 8) // row stride: 4168 elems (8336 B) -> 36-bank shift/row
#define ATOTAL  (16 * RSTRIDE)  // 66,688 elems = 133,376 B

static __device__ __forceinline__ unsigned short f2bf_rne(float f) {
  unsigned u = __float_as_uint(f);
  unsigned r = u + 0x7FFFu + ((u >> 16) & 1u);   // round-to-nearest-even
  return (unsigned short)(r >> 16);
}
static __device__ __forceinline__ float bf2f(unsigned short h) {
  return __uint_as_float(((unsigned)h) << 16);
}

// ---------------------------------------------------------------------------
// Phase 0: split fp32 embeddings into bf16 hi/lo pair, compute row sq-norms.
// ---------------------------------------------------------------------------
__global__ void __launch_bounds__(256)
triplet_prep(const float* __restrict__ X,
             unsigned short* __restrict__ Xhi,
             unsigned short* __restrict__ Xlo,
             float* __restrict__ sq) {
  const int lane = threadIdx.x & 31;
  const int wave = threadIdx.x >> 5;
  const int row  = blockIdx.x * 8 + wave;
  const size_t base = (size_t)row * DCONST;

  float s = 0.0f;
#pragma unroll
  for (int i = 0; i < DCONST / 128; ++i) {
    const int idx = i * 128 + lane * 4;
    f4 x = *(const f4*)(X + base + idx);
    us4 hi, lo;
#pragma unroll
    for (int j = 0; j < 4; ++j) {
      float xv = x[j];
      unsigned short h = f2bf_rne(xv);
      float hf = bf2f(h);
      unsigned short l = f2bf_rne(xv - hf);
      hi[j] = h; lo[j] = l;
      s += xv * xv;
    }
    *(us4*)(Xhi + base + idx) = hi;
    *(us4*)(Xlo + base + idx) = lo;
  }
#pragma unroll
  for (int off = 16; off > 0; off >>= 1) s += __shfl_xor(s, off, 32);
  if (lane == 0) sq[row] = s;
}

// ---------------------------------------------------------------------------
// Phase 1: fused Gram + hardest-pos/neg mining, M=64 blocking.
// ---------------------------------------------------------------------------
__global__ void __launch_bounds__(256)
triplet_hardmine(const unsigned short* __restrict__ Xhi,
                 const unsigned short* __restrict__ Xlo,
                 const float* __restrict__ sq,
                 const int* __restrict__ labels,
                 float* __restrict__ hpOut,
                 float* __restrict__ hnOut,
                 int B) {
  __shared__ unsigned short Asm[ATOTAL];   // 133,376 B
  __shared__ float rowInfo[MROWS * 2];     // (sq, label) pairs, 512 B
  __shared__ float redHp[8][8][32];        // 8 KB (reused per M-round)
  __shared__ float redHn[8][8][32];        // 8 KB

  const int tid     = threadIdx.x;
  const int lane    = tid & 31;
  const int wave    = tid >> 5;
  const int rowbase = blockIdx.x * MROWS;

  // Stage (sq, label) pairs for the 64 block rows.
  if (tid < MROWS) {
    rowInfo[2 * tid]     = sq[rowbase + tid];
    rowInfo[2 * tid + 1] = __int_as_float(labels[rowbase + tid]);
  }
  // Cooperative stage of the 64-row A block into [row][part][m][KPAD].
  for (int idx = tid; idx < MROWS * (DCONST / 8); idx += 256) {
    const int r   = idx >> 6;            // block row 0..63
    const int seg = idx & 63;            // 8-elem segment 0..63
    const int m   = r >> 4;
    const int rr  = r & 15;
    const size_t g = (size_t)(rowbase + r) * DCONST + seg * 8;
    *(i4*)&Asm[rr * RSTRIDE + m * KPAD + seg * 8]           = *(const i4*)&Xhi[g];
    *(i4*)&Asm[rr * RSTRIDE + PSTRIDE + m * KPAD + seg * 8] = *(const i4*)&Xlo[g];
  }
  __syncthreads();

  const int lrow = lane & 15;            // row (A) / col (B) within 16-group
  const int koff = (lane >> 4) * 8;      // ISA 16-bit operand K striping
  const int mrow = (lane >> 4) * 8;      // C/D: VGPR j holds M = j + mrow

  // One per-lane LDS base; every A access below is base + compile-time const.
  const unsigned short* pA = &Asm[lrow * RSTRIDE + koff];

  float hp[MTILES][8], hn[MTILES][8];
#pragma unroll
  for (int m = 0; m < MTILES; ++m)
#pragma unroll
    for (int j = 0; j < 8; ++j) { hp[m][j] = -BIGV; hn[m][j] = BIGV; }

  const int ntiles = B >> 4;
  for (int t = wave; t < ntiles; t += 8) {
    const int c     = t * 16 + lrow;
    const float sqc = sq[c];
    const int  labc = labels[c];
    const unsigned short* Bhp = Xhi + (size_t)c * DCONST;
    const unsigned short* Blp = Xlo + (size_t)c * DCONST;

    v8f acc[MTILES];
#pragma unroll
    for (int m = 0; m < MTILES; ++m) { v8f z = {}; acc[m] = z; }

#pragma unroll 2
    for (int kb = 0; kb < DCONST; kb += 32) {
      const int k0 = kb + koff;
      // B operand from global/L2: reused by all 4 M-subtiles.
      v8bf bh0 = *(const v8bf*)(Bhp + k0);
      v8bf bh1 = *(const v8bf*)(Bhp + k0 + 16);
      v8bf bl0 = *(const v8bf*)(Blp + k0);
      v8bf bl1 = *(const v8bf*)(Blp + k0 + 16);
      v16bf bh = __builtin_shufflevector(bh0, bh1, 0,1,2,3,4,5,6,7,8,9,10,11,12,13,14,15);
      v16bf bl = __builtin_shufflevector(bl0, bl1, 0,1,2,3,4,5,6,7,8,9,10,11,12,13,14,15);

      // A-hi operands (immediate-offset DS loads), used by terms 1 & 2.
      v16bf ah[MTILES];
#pragma unroll
      for (int m = 0; m < MTILES; ++m) {
        v8bf h0 = *(const v8bf*)(pA + m * KPAD + kb);
        v8bf h1 = *(const v8bf*)(pA + m * KPAD + kb + 16);
        ah[m] = __builtin_shufflevector(h0, h1, 0,1,2,3,4,5,6,7,8,9,10,11,12,13,14,15);
      }
#pragma unroll
      for (int m = 0; m < MTILES; ++m)
        acc[m] = __builtin_amdgcn_wmma_f32_16x16x32_bf16(false, ah[m], false, bh, (short)0, acc[m], false, false);
#pragma unroll
      for (int m = 0; m < MTILES; ++m)
        acc[m] = __builtin_amdgcn_wmma_f32_16x16x32_bf16(false, ah[m], false, bl, (short)0, acc[m], false, false);

      // A-lo operands loaded late: live only across term 3.
      v16bf al[MTILES];
#pragma unroll
      for (int m = 0; m < MTILES; ++m) {
        v8bf l0 = *(const v8bf*)(pA + PSTRIDE + m * KPAD + kb);
        v8bf l1 = *(const v8bf*)(pA + PSTRIDE + m * KPAD + kb + 16);
        al[m] = __builtin_shufflevector(l0, l1, 0,1,2,3,4,5,6,7,8,9,10,11,12,13,14,15);
      }
#pragma unroll
      for (int m = 0; m < MTILES; ++m)
        acc[m] = __builtin_amdgcn_wmma_f32_16x16x32_bf16(false, al[m], false, bh, (short)0, acc[m], false, false);
    }

    // Epilogue: distances + hardest-pos/neg update. Row (sq,label) come from
    // LDS as one b64 load each -- not held in registers across the sweep.
#pragma unroll
    for (int m = 0; m < MTILES; ++m)
#pragma unroll
      for (int j = 0; j < 8; ++j) {
        const int rl = m * 16 + j + mrow;           // row within block
        f2 info = *(const f2*)&rowInfo[2 * rl];
        const float sqr = info[0];
        const int  labr = __float_as_int(info[1]);
        const int  r    = rowbase + rl;
        float d2 = sqr + sqc - 2.0f * acc[m][j];
        float d  = d2 > 0.0f ? sqrtf(d2) : 0.0f;
        const bool same = (labr == labc);
        hp[m][j] = fmaxf(hp[m][j], (same && (r != c)) ? d : -BIGV);
        hn[m][j] = fminf(hn[m][j], same ? BIGV : d);
      }
  }

  // Cross-wave reduction: 4 sequential 16-row rounds through 16KB LDS.
  for (int m = 0; m < MTILES; ++m) {
    __syncthreads();
#pragma unroll
    for (int j = 0; j < 8; ++j) {
      redHp[wave][j][lane] = hp[m][j];
      redHn[wave][j][lane] = hn[m][j];
    }
    __syncthreads();
    // Row r (0..15): register j = r&7, lane half = (r<8) ? 0-15 : 16-31.
    if (tid < 16) {
      const int j  = tid & 7;
      const int l0 = (tid < 8) ? 0 : 16;
      float mh = -BIGV, mn = BIGV;
      for (int w = 0; w < 8; ++w)
#pragma unroll
        for (int l = 0; l < 16; ++l) {
          mh = fmaxf(mh, redHp[w][j][l0 + l]);
          mn = fminf(mn, redHn[w][j][l0 + l]);
        }
      hpOut[rowbase + m * 16 + tid] = mh;
      hnOut[rowbase + m * 16 + tid] = mn;
    }
  }
}

// ---------------------------------------------------------------------------
// Phase 2: mean of positive losses over valid rows.
// ---------------------------------------------------------------------------
__global__ void __launch_bounds__(256)
triplet_finalize(const float* __restrict__ hp,
                 const float* __restrict__ hn,
                 float* __restrict__ out, int B) {
  __shared__ float ssum[256];
  __shared__ float scnt[256];
  const int tid = threadIdx.x;
  float s = 0.0f, c = 0.0f;
  for (int i = tid; i < B; i += 256) {
    const float h = hp[i], n = hn[i];
    const float l = h - n + MARGINV;
    if (h > -1.0e8f && n < 1.0e8f && l > 0.0f) { s += l; c += 1.0f; }
  }
  ssum[tid] = s; scnt[tid] = c;
  __syncthreads();
  for (int off = 128; off > 0; off >>= 1) {
    if (tid < off) { ssum[tid] += ssum[tid + off]; scnt[tid] += scnt[tid + off]; }
    __syncthreads();
  }
  if (tid == 0) out[0] = (scnt[0] > 0.0f) ? (ssum[0] / scnt[0]) : 0.0f;
}

// ---------------------------------------------------------------------------
extern "C" void kernel_launch(void* const* d_in, const int* in_sizes, int n_in,
                              void* d_out, int out_size, void* d_ws, size_t ws_size,
                              hipStream_t stream) {
  (void)n_in; (void)out_size; (void)ws_size;
  const float* X      = (const float*)d_in[0];
  const int*   labels = (const int*)d_in[1];
  const int B = in_sizes[1];                 // 4096
  // D is fixed at 512 (DCONST) per the reference problem size.

  // Workspace layout: Xhi | Xlo | sq | hp | hn   (~8.05 MB)
  unsigned short* Xhi = (unsigned short*)d_ws;
  unsigned short* Xlo = Xhi + (size_t)B * DCONST;
  float* sq = (float*)(Xlo + (size_t)B * DCONST);
  float* hp = sq + B;
  float* hn = hp + B;

  float* out = (float*)d_out;

  triplet_prep<<<B / 8, 256, 0, stream>>>(X, Xhi, Xlo, sq);
  triplet_hardmine<<<B / MROWS, 256, 0, stream>>>(Xhi, Xlo, sq, labels, hp, hn, B);
  triplet_finalize<<<1, 256, 0, stream>>>(hp, hn, out, B);
}
